// MultiLoRAMultiheadSelfAttenion_1889785611044
// MI455X (gfx1250) — matrix operand order repaired and checked by reading, not compiled, not run on hardware
//
#include <hip/hip_runtime.h>
#include <stdint.h>

#define BB 16
#define SS 2048
#define DD 1024
#define RR 16

typedef __attribute__((ext_vector_type(8)))  float        v8f;
typedef __attribute__((ext_vector_type(8)))  unsigned int v8u;
typedef __attribute__((ext_vector_type(16))) __bf16       v16bf;
typedef __attribute__((ext_vector_type(2)))  __bf16       bf16x2;
typedef __attribute__((ext_vector_type(2)))  float        f32x2;
typedef __attribute__((ext_vector_type(4)))  int          v4i;

#if __has_builtin(__builtin_amdgcn_global_load_async_to_lds_b128)
#define HAVE_ASYNC_LDS 1
#else
#define HAVE_ASYNC_LDS 0
#endif

#if HAVE_ASYNC_LDS
typedef __attribute__((address_space(1))) v4i* gv4p;   // global 128-bit pointer
typedef __attribute__((address_space(3))) v4i* lv4p;   // LDS 128-bit pointer
__device__ __forceinline__ void async_copy16(const void* g, void* l) {
    __builtin_amdgcn_global_load_async_to_lds_b128((gv4p)(v4i*)g, (lv4p)(v4i*)l, 0, 0);
}
#endif

__device__ __forceinline__ void wait_async0() {
#if HAVE_ASYNC_LDS
#if __has_builtin(__builtin_amdgcn_s_wait_asynccnt)
    __builtin_amdgcn_s_wait_asynccnt(0);
#else
    asm volatile("s_wait_asynccnt 0x0" ::: "memory");
#endif
#endif
}

// ---- native bf16 conversions (backend picks v_cvt_pk_bf16_f32 etc.) ----
__device__ __forceinline__ unsigned short f2bf(float x) {
    __bf16 h = (__bf16)x;
    return __builtin_bit_cast(unsigned short, h);
}
__device__ __forceinline__ unsigned int pack2bf(float a, float b) {
    f32x2 f = {a, b};
    bf16x2 h = __builtin_convertvector(f, bf16x2);
    return __builtin_bit_cast(unsigned int, h);
}

// ---- WMMA operand gathers per CDNA5 documented VGPR layouts (LDS stride 34 ushorts) ----
__device__ __forceinline__ v16bf fetchA_frag(const unsigned short* As, int tileRow, int lane) {
    int m  = tileRow + (lane & 15);
    int kh = (lane >> 4) << 3;
    v8u r;
#pragma unroll
    for (int v = 0; v < 8; ++v) {
        int kk = 2 * v + ((v >= 4) ? 8 : 0) + kh;      // A: K pairs 0..7 / 16..23 (+8 hi lanes)
        r[v] = *(const unsigned int*)&As[m * 34 + kk];
    }
    return __builtin_bit_cast(v16bf, r);
}
__device__ __forceinline__ v16bf fetchB_frag(const unsigned short* Bs, int tileCol, int lane) {
    int n  = tileCol + (lane & 15);
    int kh = (lane >> 4) << 3;
    v8u r;
#pragma unroll
    for (int v = 0; v < 8; ++v) {
        int kk = 2 * v + ((v >= 4) ? 8 : 0) + kh;      // B: K pairs 0..7 / 16..23 (+8 hi lanes)
        r[v] = *(const unsigned int*)&Bs[n * 34 + kk];
    }
    return __builtin_bit_cast(v16bf, r);
}

// =====================================================================================
// Generic 128x128-tile GEMM:  C[b][s][n] = alpha * sum_k A[b][s][k] * B'[k][n] (+bias[n])
//   TRANS_B=1: B given row-major [N][K] (weights / K-proj) ; TRANS_B=0: B row-major [K][N]
//   LORA: extra K=32 step with A=[u | 0] (bf16 [S][16]) and B=[loraB | 0] (f32 [D][16])
// bf16 row-major tiles are DMA'd straight into LDS with GLOBAL_LOAD_ASYNC_TO_LDS_B128
// (ASYNCcnt path); f32 tiles convert via packed VALU cvt.
// 256 threads = 8 waves; wave (wm,wn) owns a 32x64 block = 2x4 WMMA tiles.
// =====================================================================================
template <bool A_BF, bool B_BF, bool TRANS_B, bool LORA, bool OUT_BF>
__global__ __launch_bounds__(256) void gemm_wmma_kernel(
    const void* __restrict__ Ag, long sAb, int lda,
    const void* __restrict__ Bg, long sBb, int ldb,
    const float* __restrict__ bias,
    const unsigned short* __restrict__ uG,   // [B][S][16] bf16
    const float* __restrict__ loraBG,        // [B][D][16] f32
    void* __restrict__ Cg, long sCb, int ldc,
    int Kdim, float alpha)
{
    __shared__ unsigned short As[128 * 34];
    __shared__ unsigned short Bs[128 * 34];

    constexpr bool ASYNC_A = HAVE_ASYNC_LDS && A_BF;
    constexpr bool ASYNC_B = HAVE_ASYNC_LDS && B_BF && TRANS_B;

    const int tid  = threadIdx.x;
    const int lane = tid & 31;
    const int wave = tid >> 5;
    const int wm   = wave & 3;
    const int wn   = wave >> 2;
    const int b    = blockIdx.z;
    const int n0   = blockIdx.x * 128;
    const int s0   = blockIdx.y * 128;

    const float*          Af = (const float*)Ag          + (size_t)b * sAb;
    const unsigned short* Ab = (const unsigned short*)Ag + (size_t)b * sAb;
    const float*          Bf = (const float*)Bg          + (size_t)b * sBb;
    const unsigned short* Bb = (const unsigned short*)Bg + (size_t)b * sBb;

    v8f acc[2][4];
#pragma unroll
    for (int i = 0; i < 2; ++i)
#pragma unroll
        for (int j = 0; j < 4; ++j) acc[i][j] = {};

    for (int k0 = 0; k0 < Kdim; k0 += 32) {
        // ---- A tile: rows s0..+127, cols k0..+31 ----
        if (ASYNC_A) {
#if HAVE_ASYNC_LDS
#pragma unroll
            for (int i = 0; i < 2; ++i) {
                int id  = tid + i * 256;
                int r   = id >> 2;
                int seg = (id & 3) << 3;               // 8 ushorts = 16 bytes per chunk
                async_copy16(Ab + (size_t)(s0 + r) * lda + k0 + seg, &As[r * 34 + seg]);
            }
#endif
        } else {
#pragma unroll
            for (int i = 0; i < 4; ++i) {
                int id = tid + i * 256;
                int r  = id >> 3;
                int c  = (id & 7) << 2;
                unsigned int w0, w1;
                if (A_BF) {
                    const unsigned int* p = (const unsigned int*)(Ab + (size_t)(s0 + r) * lda + k0 + c);
                    w0 = p[0]; w1 = p[1];
                } else {
                    const float4 f = *(const float4*)(Af + (size_t)(s0 + r) * lda + k0 + c);
                    w0 = pack2bf(f.x, f.y); w1 = pack2bf(f.z, f.w);
                }
                *(unsigned int*)&As[r * 34 + c]     = w0;
                *(unsigned int*)&As[r * 34 + c + 2] = w1;
            }
        }
        // ---- B tile -> LDS layout [N][K] ----
        if (TRANS_B) {
            if (ASYNC_B) {
#if HAVE_ASYNC_LDS
#pragma unroll
                for (int i = 0; i < 2; ++i) {
                    int id  = tid + i * 256;
                    int r   = id >> 2;
                    int seg = (id & 3) << 3;
                    async_copy16(Bb + (size_t)(n0 + r) * ldb + k0 + seg, &Bs[r * 34 + seg]);
                }
#endif
            } else {
#pragma unroll
                for (int i = 0; i < 4; ++i) {
                    int id = tid + i * 256;
                    int r  = id >> 3;
                    int c  = (id & 7) << 2;
                    unsigned int w0, w1;
                    if (B_BF) {
                        const unsigned int* p = (const unsigned int*)(Bb + (size_t)(n0 + r) * ldb + k0 + c);
                        w0 = p[0]; w1 = p[1];
                    } else {
                        const float4 f = *(const float4*)(Bf + (size_t)(n0 + r) * ldb + k0 + c);
                        w0 = pack2bf(f.x, f.y); w1 = pack2bf(f.z, f.w);
                    }
                    *(unsigned int*)&Bs[r * 34 + c]     = w0;
                    *(unsigned int*)&Bs[r * 34 + c + 2] = w1;
                }
            }
        } else {
#pragma unroll
            for (int i = 0; i < 4; ++i) {
                int id = tid + i * 256;
                int k  = id >> 5;
                int n  = (id & 31) << 2;
                unsigned short e0, e1, e2, e3;
                if (B_BF) {
                    const unsigned short* p = Bb + (size_t)(k0 + k) * ldb + n0 + n;
                    e0 = p[0]; e1 = p[1]; e2 = p[2]; e3 = p[3];
                } else {
                    const float4 f = *(const float4*)(Bf + (size_t)(k0 + k) * ldb + n0 + n);
                    e0 = f2bf(f.x); e1 = f2bf(f.y); e2 = f2bf(f.z); e3 = f2bf(f.w);
                }
                Bs[(n + 0) * 34 + k] = e0;
                Bs[(n + 1) * 34 + k] = e1;
                Bs[(n + 2) * 34 + k] = e2;
                Bs[(n + 3) * 34 + k] = e3;
            }
        }
        if (ASYNC_A || ASYNC_B) wait_async0();
        __syncthreads();

        v16bf af[2], bfr[4];
#pragma unroll
        for (int mt = 0; mt < 2; ++mt) af[mt] = fetchA_frag(As, wm * 32 + mt * 16, lane);
#pragma unroll
        for (int nt = 0; nt < 4; ++nt) bfr[nt] = fetchB_frag(Bs, wn * 64 + nt * 16, lane);
#pragma unroll
        for (int mt = 0; mt < 2; ++mt)
#pragma unroll
            for (int nt = 0; nt < 4; ++nt)
                acc[mt][nt] = __builtin_amdgcn_wmma_f32_16x16x32_bf16(
                    false, af[mt], false, bfr[nt], (short)0, acc[mt][nt], false, false);
        __syncthreads();
    }

    if (LORA) {
        // A = [u(16) | zeros(16)]
        {
            int r = tid >> 1;
            int h = (tid & 1) << 3;
            const unsigned int* p =
                (const unsigned int*)(uG + (size_t)b * SS * 16 + (size_t)(s0 + r) * 16 + h);
            *(unsigned int*)&As[r * 34 + h + 0] = p[0];
            *(unsigned int*)&As[r * 34 + h + 2] = p[1];
            *(unsigned int*)&As[r * 34 + h + 4] = p[2];
            *(unsigned int*)&As[r * 34 + h + 6] = p[3];
            *(unsigned int*)&As[r * 34 + 16 + h + 0] = 0u;
            *(unsigned int*)&As[r * 34 + 16 + h + 2] = 0u;
            *(unsigned int*)&As[r * 34 + 16 + h + 4] = 0u;
            *(unsigned int*)&As[r * 34 + 16 + h + 6] = 0u;
        }
        // B = [loraB(16) | zeros(16)]
#pragma unroll
        for (int i = 0; i < 2; ++i) {
            int id = tid + i * 256;
            int r  = id >> 2;
            int c  = (id & 3) << 2;
            const float4 f =
                *(const float4*)(loraBG + (size_t)b * DD * 16 + (size_t)(n0 + r) * 16 + c);
            *(unsigned int*)&Bs[r * 34 + c]          = pack2bf(f.x, f.y);
            *(unsigned int*)&Bs[r * 34 + c + 2]      = pack2bf(f.z, f.w);
            *(unsigned int*)&Bs[r * 34 + 16 + c]     = 0u;
            *(unsigned int*)&Bs[r * 34 + 16 + c + 2] = 0u;
        }
        __syncthreads();
        v16bf af[2], bfr[4];
#pragma unroll
        for (int mt = 0; mt < 2; ++mt) af[mt] = fetchA_frag(As, wm * 32 + mt * 16, lane);
#pragma unroll
        for (int nt = 0; nt < 4; ++nt) bfr[nt] = fetchB_frag(Bs, wn * 64 + nt * 16, lane);
#pragma unroll
        for (int mt = 0; mt < 2; ++mt)
#pragma unroll
            for (int nt = 0; nt < 4; ++nt)
                acc[mt][nt] = __builtin_amdgcn_wmma_f32_16x16x32_bf16(
                    false, af[mt], false, bfr[nt], (short)0, acc[mt][nt], false, false);
    }

    // ---- store (C/D layout: VGPR v -> row v + 8*(lane>=16), col = lane&15) ----
    float*          Cf = (float*)Cg          + (size_t)b * sCb;
    unsigned short* Cb = (unsigned short*)Cg + (size_t)b * sCb;
    const int mo = (lane >> 4) << 3;
#pragma unroll
    for (int mt = 0; mt < 2; ++mt) {
#pragma unroll
        for (int nt = 0; nt < 4; ++nt) {
            int gn   = n0 + wn * 64 + nt * 16 + (lane & 15);
            float bv = bias ? bias[gn] : 0.f;
            int gm0  = s0 + wm * 32 + mt * 16 + mo;
#pragma unroll
            for (int v = 0; v < 8; ++v) {
                float val = acc[mt][nt][v] * alpha + bv;
                if (OUT_BF) Cb[(size_t)(gm0 + v) * ldc + gn] = f2bf(val);
                else        Cf[(size_t)(gm0 + v) * ldc + gn] = val;
            }
        }
    }
}

// =============================  u = query @ lora_A^T  (N=16)  ========================
__global__ __launch_bounds__(256) void lora_u_kernel(const float* __restrict__ query,
                                                     const float* __restrict__ loraA,
                                                     unsigned short* __restrict__ uG)
{
    __shared__ unsigned short As[128 * 34];
    __shared__ unsigned short Bs[16 * 34];
    const int tid  = threadIdx.x;
    const int lane = tid & 31;
    const int wave = tid >> 5;
    const int b    = blockIdx.y;
    const int s0   = blockIdx.x * 128;
    const float* Af = query + (size_t)b * SS * DD;
    const float* Bf = loraA + (size_t)b * RR * DD;

    v8f acc = {};
    for (int k0 = 0; k0 < DD; k0 += 32) {
#pragma unroll
        for (int i = 0; i < 4; ++i) {
            int id = tid + i * 256;
            int r  = id >> 3;
            int c  = (id & 7) << 2;
            const float4 f = *(const float4*)(Af + (size_t)(s0 + r) * DD + k0 + c);
            *(unsigned int*)&As[r * 34 + c]     = pack2bf(f.x, f.y);
            *(unsigned int*)&As[r * 34 + c + 2] = pack2bf(f.z, f.w);
        }
        if (tid < 128) {
            int r = tid >> 3;
            int c = (tid & 7) << 2;
            const float4 f = *(const float4*)(Bf + (size_t)r * DD + k0 + c);
            *(unsigned int*)&Bs[r * 34 + c]     = pack2bf(f.x, f.y);
            *(unsigned int*)&Bs[r * 34 + c + 2] = pack2bf(f.z, f.w);
        }
        __syncthreads();
        v16bf a  = fetchA_frag(As, wave * 16, lane);
        v16bf bb = fetchB_frag(Bs, 0, lane);
        acc = __builtin_amdgcn_wmma_f32_16x16x32_bf16(false, a, false, bb, (short)0, acc,
                                                      false, false);
        __syncthreads();
    }
    int gn  = lane & 15;
    int gm0 = s0 + wave * 16 + ((lane >> 4) << 3);
#pragma unroll
    for (int v = 0; v < 8; ++v)
        uG[(size_t)b * SS * 16 + (size_t)(gm0 + v) * 16 + gn] = f2bf(acc[v]);
}

// ====================  row softmax in place, applies 1/sqrt(D) scale  ================
__global__ __launch_bounds__(256) void softmax_kernel(float* __restrict__ scores)
{
    const int q = blockIdx.x;
    const int b = blockIdx.y;
    float* row  = scores + ((size_t)b * SS + q) * SS;
    __shared__ float red[256];
    const int t = threadIdx.x;

    float mx = -3.4e38f;
    for (int i = t; i < SS; i += 256) mx = fmaxf(mx, row[i]);
    red[t] = mx; __syncthreads();
    for (int off = 128; off; off >>= 1) { if (t < off) red[t] = fmaxf(red[t], red[t + off]); __syncthreads(); }
    mx = red[0]; __syncthreads();

    const float scale = 0.03125f;   // 1/sqrt(1024)
    float sum = 0.f;
    for (int i = t; i < SS; i += 256) {
        float e = __expf((row[i] - mx) * scale);
        row[i]  = e;
        sum    += e;
    }
    red[t] = sum; __syncthreads();
    for (int off = 128; off; off >>= 1) { if (t < off) red[t] += red[t + off]; __syncthreads(); }
    const float inv = 1.0f / red[0];
    for (int i = t; i < SS; i += 256) row[i] *= inv;
}

extern "C" void kernel_launch(void* const* d_in, const int* in_sizes, int n_in,
                              void* d_out, int out_size, void* d_ws, size_t ws_size,
                              hipStream_t stream)
{
    (void)in_sizes; (void)n_in; (void)out_size; (void)ws_size;
    const float* query = (const float*)d_in[0];
    const float* key   = (const float*)d_in[1];
    /* d_in[2] = value : unused — reference computes V from `key` */
    const float* Wq = (const float*)d_in[3];
    const float* bq = (const float*)d_in[4];
    const float* Wk = (const float*)d_in[5];
    const float* bk = (const float*)d_in[6];
    const float* Wv = (const float*)d_in[7];
    const float* bv = (const float*)d_in[8];
    const float* loraA = (const float*)d_in[9];
    const float* loraB = (const float*)d_in[10];

    char* ws = (char*)d_ws;
    const size_t qkvBytes = (size_t)BB * SS * DD * 2;       // 64 MiB each (bf16)
    unsigned short* Qb = (unsigned short*)(ws);
    unsigned short* Kb = (unsigned short*)(ws + qkvBytes);
    unsigned short* Vb = (unsigned short*)(ws + 2 * qkvBytes);
    unsigned short* Ub = (unsigned short*)(ws + 3 * qkvBytes);
    float* Sc = (float*)(ws + 3 * qkvBytes + (size_t)BB * SS * RR * 2);  // [B][S][S] f32

    const long sQK = (long)SS * DD;

    // 1) u = query @ lora_A^T  -> bf16 [B][S][16]
    lora_u_kernel<<<dim3(SS / 128, BB), 256, 0, stream>>>(query, loraA, Ub);

    // 2) Q = query @ Wq^T + u @ loraB^T + bq   (bf16)
    gemm_wmma_kernel<false, false, true, true, true>
        <<<dim3(DD / 128, SS / 128, BB), 256, 0, stream>>>(
            query, sQK, DD, Wq, 0L, DD, bq, Ub, loraB, Qb, sQK, DD, DD, 1.0f);

    // 3) K = key @ Wk^T + bk   (bf16)
    gemm_wmma_kernel<false, false, true, false, true>
        <<<dim3(DD / 128, SS / 128, BB), 256, 0, stream>>>(
            key, sQK, DD, Wk, 0L, DD, bk, nullptr, nullptr, Kb, sQK, DD, DD, 1.0f);

    // 4) V = key @ Wv^T + bv   (bf16)
    gemm_wmma_kernel<false, false, true, false, true>
        <<<dim3(DD / 128, SS / 128, BB), 256, 0, stream>>>(
            key, sQK, DD, Wv, 0L, DD, bv, nullptr, nullptr, Vb, sQK, DD, DD, 1.0f);

    // 5) scores = Q @ K^T  (f32, raw; scale folded into softmax) — async-LDS tile feed
    gemm_wmma_kernel<true, true, true, false, false>
        <<<dim3(SS / 128, SS / 128, BB), 256, 0, stream>>>(
            Qb, sQK, DD, Kb, sQK, DD, nullptr, nullptr, nullptr,
            Sc, (long)SS * SS, SS, DD, 1.0f);

    // 6) softmax rows in place (applies 1/32 scale)
    softmax_kernel<<<dim3(SS, BB), 256, 0, stream>>>(Sc);

    // 7) out = P @ V   (f32)
    gemm_wmma_kernel<false, true, false, false, false>
        <<<dim3(DD / 128, SS / 128, BB), 256, 0, stream>>>(
            Sc, (long)SS * SS, SS, Vb, sQK, DD, nullptr, nullptr, nullptr,
            (float*)d_out, sQK, DD, SS, 1.0f);
}
